// nms_26989574488094
// MI455X (gfx1250) — 8-bit matrix operand order repaired and checked by reading, compiled, not run on hardware
//
#include <hip/hip_runtime.h>

typedef int v8i __attribute__((ext_vector_type(8)));
typedef unsigned int u32;

#define N_INST 64
#define CHUNK  2048
#define SDW    ((CHUNK + 16) / 4)
#define THREADS 512

__global__ void zero_gram_kernel(int* gram) {
    int i = blockIdx.x * blockDim.x + threadIdx.x;
    if (i < N_INST * N_INST) gram[i] = 0;
}

__global__ __launch_bounds__(THREADS)
void gram_iu8_kernel(const float* __restrict__ mask, int* __restrict__ gram, long long V) {
    extern __shared__ u32 lds[];
    const int tid  = threadIdx.x;
    const int lane = tid & 31;
    const int wave = tid >> 5;
    const int ti   = wave >> 2;
    const int tj   = wave & 3;
    const int hl   = lane >> 4;
    const int lrow = lane & 15;

    v8i c = {};

    const long long nchunks = (V + CHUNK - 1) / CHUNK;
    for (long long ch = blockIdx.x; ch < nchunks; ch += gridDim.x) {
        const long long k0 = ch * (long long)CHUNK;

        for (int r = 0; r < N_INST; ++r) {
            const long long kk = k0 + (long long)tid * 4;
            u32 packed = 0u;
            if (kk + 3 < V) {
                const float4 f = *(const float4*)(mask + (long long)r * V + kk);
                packed = (u32)(f.x != 0.0f)
                       | ((u32)(f.y != 0.0f) << 8)
                       | ((u32)(f.z != 0.0f) << 16)
                       | ((u32)(f.w != 0.0f) << 24);
            } else {
                for (int t = 0; t < 4; ++t) {
                    const long long k = kk + t;
                    if (k < V && mask[(long long)r * V + k] != 0.0f)
                        packed |= 1u << (8 * t);
                }
            }
            lds[r * SDW + tid] = packed;
        }
        __syncthreads();

        const u32* rowA = lds + (ti * 16 + lrow) * SDW + hl * 4;
        const u32* rowB = lds + (tj * 16 + lrow) * SDW + hl * 4;
        for (int ks = 0; ks < CHUNK; ks += 64) {
            const int kd = ks >> 2;
            v8i a, b;
            {
                uint2 p0 = *(const uint2*)(rowA + kd + 0);
                uint2 p1 = *(const uint2*)(rowA + kd + 2);
                uint2 p2 = *(const uint2*)(rowA + kd + 8);
                uint2 p3 = *(const uint2*)(rowA + kd + 10);
                a[0] = (int)p0.x; a[1] = (int)p0.y; a[2] = (int)p1.x; a[3] = (int)p1.y;
                a[4] = (int)p2.x; a[5] = (int)p2.y; a[6] = (int)p3.x; a[7] = (int)p3.y;
            }
            {
                uint4 q0 = *(const uint4*)(rowB + kd + 0);
                uint4 q1 = *(const uint4*)(rowB + kd + 8);
                b[0] = (int)q0.x; b[1] = (int)q0.y; b[2] = (int)q0.z; b[3] = (int)q0.w;
                b[4] = (int)q1.x; b[5] = (int)q1.y; b[6] = (int)q1.z; b[7] = (int)q1.w;
            }
            c = __builtin_amdgcn_wmma_i32_16x16x64_iu8(false, a, false, b, c, false, false);
        }
        __syncthreads();
    }

    const int col = tj * 16 + lrow;
    for (int r = 0; r < 8; ++r) {
        const int row = ti * 16 + hl * 8 + r;
        atomicAdd(&gram[row * N_INST + col], c[r]);
    }
}

__global__ void nms_suppress_kernel(const int* __restrict__ gram, float* __restrict__ out) {
    if (threadIdx.x != 0 || blockIdx.x != 0) return;
    int ind[N_INST];
    for (int i = 0; i < N_INST; ++i) ind[i] = 1;
    for (int i = 0; i < N_INST; ++i) {
        if (ind[i] == 0) continue;
        const long long si = gram[i * N_INST + i];
        for (int j = 0; j < N_INST; ++j) {
            const long long inter = gram[i * N_INST + j];
            const long long sj    = gram[j * N_INST + j];
            const bool hit = 3 * inter > si + sj;
            if (hit && sj > si) ind[i] = 0;
            if (hit && si > sj) { ind[j] = 0; break; }
        }
    }
    for (int i = 0; i < N_INST; ++i) out[i] = ind[i] ? 1.0f : 0.0f;
}

extern "C" void kernel_launch(void* const* d_in, const int* in_sizes, int n_in,
                              void* d_out, int out_size, void* d_ws, size_t ws_size,
                              hipStream_t stream) {
    const float* mask = (const float*)d_in[0];
    int*   gram = (int*)d_ws;
    float* out  = (float*)d_out;
    const long long V = (long long)in_sizes[0] / N_INST;

    zero_gram_kernel<<<(N_INST * N_INST + 255) / 256, 256, 0, stream>>>(gram);

    const size_t ldsBytes = (size_t)N_INST * SDW * sizeof(u32);
    gram_iu8_kernel<<<256, THREADS, ldsBytes, stream>>>(mask, gram, V);

    nms_suppress_kernel<<<1, 32, 0, stream>>>(gram, out);
}
